// STFT_78022375899170
// MI455X (gfx1250) — matrix operand order repaired and checked by reading, not compiled, not run on hardware
//
#include <hip/hip_runtime.h>

typedef __attribute__((ext_vector_type(16))) _Float16 v16h;
typedef __attribute__((ext_vector_type(8)))  float    v8f;
typedef int v4i __attribute__((vector_size(16)));

#if __has_builtin(__builtin_amdgcn_global_load_async_to_lds_b128)
#define USE_ASYNC 1
#else
#define USE_ASYNC 0
#endif

#define XLEN   2097152L          // samples per batch
#define XPLEN  2098176L          // XLEN + 2*512 (reflect pad)
#define NFRAMES 4097
#define NBATCH  8
#define KDIM    1024
#define OUTHALF (8L * 4097L * 512L)

#define BM   128                 // frames per block tile
#define BN   128                 // output cols per block tile
#define KCH  32                  // K per chunk (one wmma depth)
#define STRB 80                  // LDS row stride bytes: 32 halves (64B) + 16B pad
#define ABUF (BM * STRB)         // 10240 B
#define BBUF (BN * STRB)         // 10240 B
#define BUFSZ (ABUF + BBUF)      // 20480 B, double buffered -> 40960 B

#if USE_ASYNC
#define GLOBAL_AS __attribute__((address_space(1)))
#define LDS_AS    __attribute__((address_space(3)))
typedef GLOBAL_AS v4i* gp_v4i;
typedef LDS_AS    v4i* lp_v4i;
static __device__ __forceinline__ void async_cp16_0(const void* g, void* l) {
  __builtin_amdgcn_global_load_async_to_lds_b128((gp_v4i)g, (lp_v4i)l, 0, 0);
}
static __device__ __forceinline__ void async_cp16_16(const void* g, void* l) {
  __builtin_amdgcn_global_load_async_to_lds_b128((gp_v4i)g, (lp_v4i)l, 16, 0);
}
static __device__ __forceinline__ void wait_async0() {
#if __has_builtin(__builtin_amdgcn_s_wait_asynccnt)
  __builtin_amdgcn_s_wait_asynccnt(0);
#else
  asm volatile("s_wait_asynccnt 0" ::: "memory");
#endif
}
#endif

// ---------------------------------------------------------------------------
// Stage 1a: reflect-pad x and convert to f16 into ws.
// ---------------------------------------------------------------------------
__global__ __launch_bounds__(256) void prep_xp(const float* __restrict__ x,
                                               _Float16* __restrict__ xp) {
  const int b = blockIdx.y;
  const long i0 = (((long)blockIdx.x * blockDim.x) + threadIdx.x) * 8L;
  if (i0 >= XPLEN) return;
  const float* xb = x + (size_t)b * XLEN;
  union { _Float16 h[8]; uint4 q; } u;
#pragma unroll
  for (int j = 0; j < 8; ++j) {
    long s = i0 + j - 512L;
    if (s < 0) s = -s;
    else if (s >= XLEN) s = 2L * XLEN - 2L - s;
    u.h[j] = (_Float16)xb[s];
  }
  *(uint4*)(xp + (size_t)b * XPLEN + i0) = u.q;
}

// ---------------------------------------------------------------------------
// Stage 1b: gather + transpose basis columns to f16, N-major.
// ---------------------------------------------------------------------------
__global__ __launch_bounds__(256) void prep_basis(const float* __restrict__ basis,
                                                  _Float16* __restrict__ Bt) {
  const int k = blockIdx.x * blockDim.x + threadIdx.x;  // 0..1023
  const int c = blockIdx.y;                             // 0..1023
  const int src = (c < 512) ? (c + 1) : (c + 2);
  Bt[(size_t)c * KDIM + k] = (_Float16)basis[(size_t)k * 1026 + src];
}

// ---------------------------------------------------------------------------
// Stage 2: tiled WMMA GEMM.  grid = (8 N-tiles, 33 M-tiles, 8 batches)
// 8 waves per block, 2(M) x 4(N); each wave: 64x32 = 4x2 WMMA subtiles.
// ---------------------------------------------------------------------------
__global__ __launch_bounds__(256)
__attribute__((amdgpu_waves_per_eu(2, 8)))
void stft_gemm(const _Float16* __restrict__ xp,
               const _Float16* __restrict__ Bt,
               float* __restrict__ out) {
  __shared__ unsigned char smem[2 * BUFSZ];

  const int tid    = threadIdx.x;
  const int lane   = tid & 31;
  const int waveId = tid >> 5;
  const int wm     = waveId & 1;    // 0..1  -> M offset wm*64
  const int wn     = waveId >> 1;   // 0..3  -> N offset wn*32
  const int ln15   = lane & 15;
  const int hiK    = lane >> 4;

  const int batch = blockIdx.z;
  const int mtile = blockIdx.y;     // 0..32
  const int ntile = blockIdx.x;     // 0..7

  // staging: A = one row / 16-half segment per thread; B likewise.
  const int aRow = tid >> 1;
  const int aSeg = tid & 1;
  int aFrame = mtile * BM + aRow;
  if (aFrame > NFRAMES - 1) aFrame = NFRAMES - 1;
  const size_t aG = (size_t)batch * XPLEN + (size_t)aFrame * 512;
  const int bCol = tid >> 1;
  const int bSeg = tid & 1;
  const int bColG = ntile * BN + bCol;
  const size_t bG = (size_t)bColG * KDIM;

  const _Float16* aP0 = xp + aG + (size_t)aSeg * 16;  // + k0
  const _Float16* bP0 = Bt + bG + (size_t)bSeg * 16;  // + k0
  unsigned char* aL0 = smem + aRow * STRB + aSeg * 32;          // + buf*BUFSZ
  unsigned char* bL0 = smem + ABUF + bCol * STRB + bSeg * 32;   // + buf*BUFSZ

  v8f acc[4][2];
#pragma unroll
  for (int i = 0; i < 4; ++i)
#pragma unroll
    for (int j = 0; j < 2; ++j) acc[i][j] = {};

  // stage chunk (k0 = kt*KCH) into buffer `buf`.
  auto stage = [&](int buf, int k0) {
#if USE_ASYNC
    async_cp16_0 (aP0 + k0, aL0 + buf * BUFSZ);
    async_cp16_16(aP0 + k0, aL0 + buf * BUFSZ);
    async_cp16_0 (bP0 + k0, bL0 + buf * BUFSZ);
    async_cp16_16(bP0 + k0, bL0 + buf * BUFSZ);
#else
    const uint4* ap = (const uint4*)(aP0 + k0);
    const uint4* bp = (const uint4*)(bP0 + k0);
    uint4 a0 = ap[0], a1 = ap[1], b0 = bp[0], b1 = bp[1];
    uint4* ad = (uint4*)(aL0 + buf * BUFSZ);
    uint4* bd = (uint4*)(bL0 + buf * BUFSZ);
    ad[0] = a0; ad[1] = a1; bd[0] = b0; bd[1] = b1;
#endif
  };

  // A fragment (16-bit 16x32 map): lane<16 -> M row, halves0-7=K0-7, 8-15=K16-23;
  // lane>=16 -> K+8 / K+24.
  auto fragA = [&](int buf, int mi) -> v16h {
    const unsigned char* base =
        smem + buf * BUFSZ + (wm * 64 + mi * 16 + ln15) * STRB + hiK * 16;
    union { uint4 q[2]; v16h h; } u;
    u.q[0] = *(const uint4*)(base);
    u.q[1] = *(const uint4*)(base + 32);
    return u.h;
  };
  // B fragment (16-bit 32x16 map): lane<16 -> N=lane, K0-7, K16-23; lane>=16 -> K8-15, K24-31.
  auto fragB = [&](int buf, int ni) -> v16h {
    const unsigned char* base =
        smem + buf * BUFSZ + ABUF + (wn * 32 + ni * 16 + ln15) * STRB + hiK * 16;
    union { uint4 q[2]; v16h h; } u;
    u.q[0] = *(const uint4*)(base);
    u.q[1] = *(const uint4*)(base + 32);
    return u.h;
  };

  // ---- prologue ----
  stage(0, 0);
#if USE_ASYNC
  wait_async0();
#endif
  __syncthreads();

  // ---- K loop: 32 chunks, double buffered, 1 barrier per chunk ----
  for (int kt = 0; kt < KDIM / KCH; ++kt) {
    const int cur = kt & 1;
    // buf[1-cur] became free at the last barrier: stage next chunk into it now.
    if (kt + 1 < KDIM / KCH) stage(1 - cur, (kt + 1) * KCH);

    v16h af[4], bf[2];
#pragma unroll
    for (int mi = 0; mi < 4; ++mi) af[mi] = fragA(cur, mi);
#pragma unroll
    for (int ni = 0; ni < 2; ++ni) bf[ni] = fragB(cur, ni);
#pragma unroll
    for (int mi = 0; mi < 4; ++mi)
#pragma unroll
      for (int ni = 0; ni < 2; ++ni)
        acc[mi][ni] = __builtin_amdgcn_wmma_f32_16x16x32_f16(
            false, af[mi], false, bf[ni], (short)0, acc[mi][ni], false, false);

#if USE_ASYNC
    wait_async0();          // our async writes into buf[1-cur] are complete
#endif
    __syncthreads();
  }

  // ---- epilogue: scatter C per the 16x16 f32 D layout ----
  const int mRow0 = hiK * 8;
#pragma unroll
  for (int mi = 0; mi < 4; ++mi) {
#pragma unroll
    for (int ni = 0; ni < 2; ++ni) {
      const int col  = ntile * BN + wn * 32 + ni * 16 + ln15;
      const int part = col >> 9;        // 0 = re, 1 = im
      const int f    = col & 511;
      const size_t outBase =
          (size_t)part * OUTHALF + ((size_t)batch * NFRAMES) * 512 + (size_t)f;
#pragma unroll
      for (int r = 0; r < 8; ++r) {
        const int frame = mtile * BM + wm * 64 + mi * 16 + mRow0 + r;
        if (frame < NFRAMES)
          out[outBase + (size_t)frame * 512] = acc[mi][ni][r];
      }
    }
  }
}

// ---------------------------------------------------------------------------
extern "C" void kernel_launch(void* const* d_in, const int* in_sizes, int n_in,
                              void* d_out, int out_size, void* d_ws, size_t ws_size,
                              hipStream_t stream) {
  (void)in_sizes; (void)n_in; (void)out_size; (void)ws_size;
  const float* x     = (const float*)d_in[0];   // (8, 2097152) f32
  const float* basis = (const float*)d_in[1];   // (1024, 1026) f32
  float* out = (float*)d_out;                   // re||im, each (8,4097,512) f32

  _Float16* xp = (_Float16*)d_ws;                                       // 33,570,816 B
  _Float16* Bt = (_Float16*)((char*)d_ws + (size_t)NBATCH * XPLEN * 2); // 2,097,152 B

  {
    dim3 grid((unsigned)((XPLEN / 8 + 255) / 256), NBATCH);
    prep_xp<<<grid, 256, 0, stream>>>(x, xp);
  }
  {
    dim3 grid(KDIM / 256, 1024);
    prep_basis<<<grid, 256, 0, stream>>>(basis, Bt);
  }
  {
    dim3 grid(8, 33, NBATCH);
    stft_gemm<<<grid, 256, 0, stream>>>(xp, Bt, out);
  }
}
